// TemporalHeteroGAT_56384330662060
// MI455X (gfx1250) — matrix operand order repaired and checked by reading, not compiled, not run on hardware
//
#include <hip/hip_runtime.h>
#include <hip/hip_bf16.h>

// ---------------- problem constants ----------------
#define NH 4
#define CC 64
#define HID 256
#define NT_ 100000
#define NU_ 30000
#define E_TT 200000
#define E_UU 200000
#define E_UT 100000

typedef __attribute__((ext_vector_type(16))) __bf16 v16bf;
typedef __attribute__((ext_vector_type(8)))  float  v8f;

// =====================================================================
// Weight transpose + f32->bf16:  W[K,N] (row major) -> Bt[N,K] bf16
// =====================================================================
__global__ void w_to_bf16_t(const float* __restrict__ W, __bf16* __restrict__ Bt,
                            int K, int N) {
  int idx = blockIdx.x * 256 + threadIdx.x;      // over N*K
  if (idx >= N * K) return;
  int n = idx / K, k = idx - n * K;
  Bt[idx] = (__bf16)W[(size_t)k * N + n];
}

// =====================================================================
// GEMM: C[M,N] = A[M,K](f32, cvt->bf16 in-reg) * Bt[N,K]^T + bias
// One wave computes a 16(M) x 64(N) tile via 4x v_wmma_f32_16x16x32_bf16.
// A frag: lane half 0 -> K {0..15}, half 1 -> K {16..31}, contiguous (same order as B)
// B frag: lane = column (0..15), half selects K {0..15} vs {16..31}, contiguous in [N,K].
// C/D: row = vgpr + 8*half, col = lane&15.
// =====================================================================
__global__ void gemm_wmma_bf16(const float* __restrict__ A,
                               const __bf16* __restrict__ Bt,
                               const float* __restrict__ bias,
                               float* __restrict__ Cout,
                               int M, int N, int K) {
  const int lane = threadIdx.x & 31;
  const int wave = threadIdx.x >> 5;
  const int tilesN = N >> 6;
  const int tilesM = M >> 4;
  const int tid = blockIdx.x * 8 + wave;
  if (tid >= tilesM * tilesN) return;            // wave-uniform exit
  const int tm = tid / tilesN;
  const int tn = tid - tm * tilesN;
  const int half = lane >> 4;
  const int rl = lane & 15;

  v8f acc0 = {}, acc1 = {}, acc2 = {}, acc3 = {};

  const float*  arow = A  + (size_t)(tm * 16 + rl) * K + half * 16;
  const __bf16* bp0  = Bt + (size_t)(tn * 64 +  0 + rl) * K + half * 16;
  const __bf16* bp1  = Bt + (size_t)(tn * 64 + 16 + rl) * K + half * 16;
  const __bf16* bp2  = Bt + (size_t)(tn * 64 + 32 + rl) * K + half * 16;
  const __bf16* bp3  = Bt + (size_t)(tn * 64 + 48 + rl) * K + half * 16;

  for (int k0 = 0; k0 < K; k0 += 32) {
    const float* ap = arow + k0;
    float4 a01 = *reinterpret_cast<const float4*>(ap);
    float4 a23 = *reinterpret_cast<const float4*>(ap + 4);
    float4 a45 = *reinterpret_cast<const float4*>(ap + 8);
    float4 a67 = *reinterpret_cast<const float4*>(ap + 12);
    v16bf af;
    af[0]=(__bf16)a01.x; af[1]=(__bf16)a01.y; af[2]=(__bf16)a01.z; af[3]=(__bf16)a01.w;
    af[4]=(__bf16)a23.x; af[5]=(__bf16)a23.y; af[6]=(__bf16)a23.z; af[7]=(__bf16)a23.w;
    af[8]=(__bf16)a45.x; af[9]=(__bf16)a45.y; af[10]=(__bf16)a45.z; af[11]=(__bf16)a45.w;
    af[12]=(__bf16)a67.x; af[13]=(__bf16)a67.y; af[14]=(__bf16)a67.z; af[15]=(__bf16)a67.w;

    v16bf b0 = *reinterpret_cast<const v16bf*>(bp0 + k0);
    v16bf b1 = *reinterpret_cast<const v16bf*>(bp1 + k0);
    v16bf b2 = *reinterpret_cast<const v16bf*>(bp2 + k0);
    v16bf b3 = *reinterpret_cast<const v16bf*>(bp3 + k0);

    acc0 = __builtin_amdgcn_wmma_f32_16x16x32_bf16(false, af, false, b0, (short)0, acc0, false, false);
    acc1 = __builtin_amdgcn_wmma_f32_16x16x32_bf16(false, af, false, b1, (short)0, acc1, false, false);
    acc2 = __builtin_amdgcn_wmma_f32_16x16x32_bf16(false, af, false, b2, (short)0, acc2, false, false);
    acc3 = __builtin_amdgcn_wmma_f32_16x16x32_bf16(false, af, false, b3, (short)0, acc3, false, false);
  }

  const int cb = tn * 64 + rl;
  float bv0 = 0.f, bv1 = 0.f, bv2 = 0.f, bv3 = 0.f;
  if (bias) { bv0 = bias[cb]; bv1 = bias[cb + 16]; bv2 = bias[cb + 32]; bv3 = bias[cb + 48]; }
#pragma unroll
  for (int v = 0; v < 8; ++v) {
    float* cr = Cout + (size_t)(tm * 16 + half * 8 + v) * N + cb;
    cr[0]  = acc0[v] + bv0;
    cr[16] = acc1[v] + bv1;
    cr[32] = acc2[v] + bv2;
    cr[48] = acc3[v] + bv3;
  }
}

// =====================================================================
// Per-node attention scores:  s[n,h] = sum_c hh[n,h,c] * a[h,c]
// =====================================================================
__global__ void node_scores(const float* __restrict__ hh, const float* __restrict__ a,
                            float* __restrict__ s, int Nn) {
  int idx = blockIdx.x * 256 + threadIdx.x;      // (n,h)
  if (idx >= Nn * NH) return;
  int n = idx >> 2, h = idx & 3;
  const float* hp = hh + (size_t)n * HID + h * CC;
  const float* ap = a + h * CC;
  float acc = 0.f;
#pragma unroll
  for (int c = 0; c < CC; c += 4) {
    float4 hv = *reinterpret_cast<const float4*>(hp + c);
    float4 av = *reinterpret_cast<const float4*>(ap + c);
    acc += hv.x * av.x + hv.y * av.y + hv.z * av.z + hv.w * av.w;
  }
  s[idx] = acc;
}

__global__ void fill_f32(float* __restrict__ p, float v, int n) {
  int i = blockIdx.x * 256 + threadIdx.x;
  if (i < n) p[i] = v;
}

__device__ __forceinline__ void atomicMaxF(float* addr, float val) {
  unsigned* ua = reinterpret_cast<unsigned*>(addr);
  unsigned old = *ua;
  while (__int_as_float((int)old) < val) {
    unsigned assumed = old;
    old = atomicCAS(ua, assumed, (unsigned)__float_as_int(val));
    if (old == assumed) break;
  }
}

// edges e >= E_real are virtual self-loops: src = dst = e - E_real
__global__ void edge_logit_max(const int* __restrict__ src, const int* __restrict__ dst,
                               int E_real, int E_tot,
                               const float* __restrict__ s_src, const float* __restrict__ s_dst,
                               float* __restrict__ m, float* __restrict__ eval) {
  int idx = blockIdx.x * 256 + threadIdx.x;      // (e,h)
  if (idx >= E_tot * NH) return;
  int e = idx >> 2, h = idx & 3;
  int s, d;
  if (e < E_real) { s = src[e]; d = dst[e]; } else { s = d = e - E_real; }
  float l = s_src[s * NH + h] + s_dst[d * NH + h];
  l = (l > 0.f) ? l : 0.2f * l;                  // leaky_relu(0.2)
  eval[idx] = l;
  atomicMaxF(&m[d * NH + h], l);
}

__global__ void edge_exp_sum(const int* __restrict__ dst, int E_real, int E_tot,
                             const float* __restrict__ m,
                             float* __restrict__ eval, float* __restrict__ denom) {
  int idx = blockIdx.x * 256 + threadIdx.x;
  if (idx >= E_tot * NH) return;
  int e = idx >> 2, h = idx & 3;
  int d = (e < E_real) ? dst[e] : (e - E_real);
  float ex = __expf(eval[idx] - m[d * NH + h]);
  eval[idx] = ex;
  atomicAdd(&denom[d * NH + h], ex);
}

// one block per edge, 256 threads = H*C channels
__global__ void edge_scatter(const int* __restrict__ src, const int* __restrict__ dst,
                             int E_real,
                             const float* __restrict__ eval, const float* __restrict__ denom,
                             const float* __restrict__ hs, float* __restrict__ acc) {
  int e = blockIdx.x;
  int hc = threadIdx.x;
  int h = hc >> 6;
  int s, d;
  if (e < E_real) { s = src[e]; d = dst[e]; } else { s = d = e - E_real; }
  float alpha = eval[e * NH + h] / (denom[d * NH + h] + 1e-16f);
  atomicAdd(&acc[(size_t)d * HID + hc], hs[(size_t)s * HID + hc] * alpha);
}

// h_new = (acc + b1 + b2 + b3)/3, optional relu
__global__ void finalize3(const float* __restrict__ acc,
                          const float* __restrict__ b1, const float* __restrict__ b2,
                          const float* __restrict__ b3,
                          float* __restrict__ hout, int Nn, int relu) {
  int idx = blockIdx.x * 256 + threadIdx.x;
  if (idx >= Nn * HID) return;
  int c = idx & 255;
  float v = (acc[idx] + b1[c] + b2[c] + b3[c]) * (1.f / 3.f);
  if (relu) v = v > 0.f ? v : 0.f;
  hout[idx] = v;
}

// =====================================================================
// Host side
// =====================================================================
struct HomoP { const float *W, *a_dst, *a_src, *b; };
struct BiP   { const float *Wd, *Ws, *a_dst, *a_src, *b; };
struct LayerP { HomoP tt_replied, tt_replies; BiP tu_posted, ut_posts; HomoP uu_inter, uu_interby; };

extern "C" void kernel_launch(void* const* d_in, const int* in_sizes, int n_in,
                              void* d_out, int out_size, void* d_ws, size_t ws_size,
                              hipStream_t stream) {
  (void)in_sizes; (void)n_in; (void)out_size; (void)ws_size;

  const float* x_t = (const float*)d_in[0];
  const float* x_u = (const float*)d_in[1];
  const int* src_replies = (const int*)d_in[2];
  const int* dst_replies = (const int*)d_in[3];
  const int* src_replied = (const int*)d_in[4];
  const int* dst_replied = (const int*)d_in[5];
  const int* src_posts   = (const int*)d_in[6];
  const int* dst_posts   = (const int*)d_in[7];
  const int* src_posted  = (const int*)d_in[8];
  const int* dst_posted  = (const int*)d_in[9];
  const int* src_inter   = (const int*)d_in[10];
  const int* dst_inter   = (const int*)d_in[11];
  const int* src_interby = (const int*)d_in[12];
  const int* dst_interby = (const int*)d_in[13];

  // params flattened as JAX pytree (sorted nested dict keys)
  int pi = 14;
  auto nextp = [&]() { return (const float*)d_in[pi++]; };
  LayerP L[2];
  for (int l = 0; l < 2; ++l) {
    auto homo = [&](HomoP& h){ h.W = nextp(); h.a_dst = nextp(); h.a_src = nextp(); h.b = nextp(); };
    auto bi   = [&](BiP& b){ b.Wd = nextp(); b.Ws = nextp(); b.a_dst = nextp(); b.a_src = nextp(); b.b = nextp(); };
    homo(L[l].tt_replied); homo(L[l].tt_replies);
    bi(L[l].tu_posted);    bi(L[l].ut_posts);
    homo(L[l].uu_inter);   homo(L[l].uu_interby);
  }
  const float* Wo = nextp(); const float* bo = nextp();   // out_t
  const float* Wp = nextp(); const float* bp = nextp();   // out_u
  const float* Wt = nextp(); const float* bt = nextp();   // proj_t
  const float* Wu = nextp(); const float* bu = nextp();   // proj_u

  // -------- workspace bump allocator --------
  char* ws = (char*)d_ws;
  size_t off = 0;
  auto balloc = [&](size_t bytes) -> void* {
    void* p = ws + off;
    off = (off + bytes + 255) & ~(size_t)255;
    return p;
  };
  float* h_t   = (float*)balloc((size_t)NT_ * HID * 4);
  float* h_u   = (float*)balloc((size_t)NU_ * HID * 4);
  float* acc_t = (float*)balloc((size_t)NT_ * HID * 4);
  float* acc_u = (float*)balloc((size_t)NU_ * HID * 4);
  float* hh_t  = (float*)balloc((size_t)NT_ * HID * 4);
  float* hh_u  = (float*)balloc((size_t)NU_ * HID * 4);
  __bf16* wbt0 = (__bf16*)balloc((size_t)256 * 768 * 2);
  __bf16* wbt1 = (__bf16*)balloc((size_t)256 * 768 * 2);
  float* s_src = (float*)balloc((size_t)NT_ * NH * 4);
  float* s_dst = (float*)balloc((size_t)NT_ * NH * 4);
  float* mbuf  = (float*)balloc((size_t)NT_ * NH * 4);
  float* dbuf  = (float*)balloc((size_t)NT_ * NH * 4);
  float* eval  = (float*)balloc((size_t)(E_TT + NT_) * NH * 4);

  auto cdiv = [](long a, long b) { return (unsigned)((a + b - 1) / b); };

  // transpose W[K,N] -> bf16 [N,K], then GEMM
  auto gemm = [&](const float* A, const float* W, int K, int N,
                  const float* bias, float* Cc, int M, __bf16* wbt) {
    hipLaunchKernelGGL(w_to_bf16_t, dim3(cdiv((long)N * K, 256)), dim3(256), 0, stream, W, wbt, K, N);
    long tiles = (long)(M / 16) * (N / 64);
    hipLaunchKernelGGL(gemm_wmma_bf16, dim3(cdiv(tiles, 8)), dim3(256), 0, stream,
                       A, wbt, bias, Cc, M, N, K);
  };

  // one GATConv scatter into accbuf (bias handled later in finalize3)
  auto gat = [&](const float* hs, int Ns, const float* hd, int Nd,
                 const float* a_src, const float* a_dst,
                 const int* srcE, const int* dstE, int E, int loops, float* accbuf) {
    int Etot = E + loops;
    hipLaunchKernelGGL(node_scores, dim3(cdiv((long)Ns * NH, 256)), dim3(256), 0, stream, hs, a_src, s_src, Ns);
    hipLaunchKernelGGL(node_scores, dim3(cdiv((long)Nd * NH, 256)), dim3(256), 0, stream, hd, a_dst, s_dst, Nd);
    hipLaunchKernelGGL(fill_f32, dim3(cdiv((long)Nd * NH, 256)), dim3(256), 0, stream, mbuf, -3.0e38f, Nd * NH);
    hipLaunchKernelGGL(fill_f32, dim3(cdiv((long)Nd * NH, 256)), dim3(256), 0, stream, dbuf, 0.f, Nd * NH);
    hipLaunchKernelGGL(edge_logit_max, dim3(cdiv((long)Etot * NH, 256)), dim3(256), 0, stream,
                       srcE, dstE, E, Etot, s_src, s_dst, mbuf, eval);
    hipLaunchKernelGGL(edge_exp_sum, dim3(cdiv((long)Etot * NH, 256)), dim3(256), 0, stream,
                       dstE, E, Etot, mbuf, eval, dbuf);
    hipLaunchKernelGGL(edge_scatter, dim3(Etot), dim3(256), 0, stream,
                       srcE, dstE, E, eval, dbuf, hs, accbuf);
  };

  // -------- input projections --------
  gemm(x_t, Wt, 768, HID, bt, h_t, NT_, wbt0);
  gemm(x_u, Wu, 128, HID, bu, h_u, NU_, wbt0);

  // -------- 2 hetero GAT layers --------
  for (int l = 0; l < 2; ++l) {
    const LayerP& P = L[l];
    hipLaunchKernelGGL(fill_f32, dim3(cdiv((long)NT_ * HID, 256)), dim3(256), 0, stream, acc_t, 0.f, NT_ * HID);
    hipLaunchKernelGGL(fill_f32, dim3(cdiv((long)NU_ * HID, 256)), dim3(256), 0, stream, acc_u, 0.f, NU_ * HID);

    // o1: tt_replies (homo, self loops)
    gemm(h_t, P.tt_replies.W, HID, HID, nullptr, hh_t, NT_, wbt0);
    gat(hh_t, NT_, hh_t, NT_, P.tt_replies.a_src, P.tt_replies.a_dst,
        src_replies, dst_replies, E_TT, NT_, acc_t);
    // o2: tt_replied
    gemm(h_t, P.tt_replied.W, HID, HID, nullptr, hh_t, NT_, wbt0);
    gat(hh_t, NT_, hh_t, NT_, P.tt_replied.a_src, P.tt_replied.a_dst,
        src_replied, dst_replied, E_TT, NT_, acc_t);
    // o3: ut_posts (src=user, dst=tweet, no loops)
    gemm(h_u, P.ut_posts.Ws, HID, HID, nullptr, hh_u, NU_, wbt0);
    gemm(h_t, P.ut_posts.Wd, HID, HID, nullptr, hh_t, NT_, wbt1);
    gat(hh_u, NU_, hh_t, NT_, P.ut_posts.a_src, P.ut_posts.a_dst,
        src_posts, dst_posts, E_UT, 0, acc_t);
    // o4: tu_posted (src=tweet, dst=user, no loops)
    gemm(h_t, P.tu_posted.Ws, HID, HID, nullptr, hh_t, NT_, wbt0);
    gemm(h_u, P.tu_posted.Wd, HID, HID, nullptr, hh_u, NU_, wbt1);
    gat(hh_t, NT_, hh_u, NU_, P.tu_posted.a_src, P.tu_posted.a_dst,
        src_posted, dst_posted, E_UT, 0, acc_u);
    // o5: uu_inter (homo, self loops)
    gemm(h_u, P.uu_inter.W, HID, HID, nullptr, hh_u, NU_, wbt0);
    gat(hh_u, NU_, hh_u, NU_, P.uu_inter.a_src, P.uu_inter.a_dst,
        src_inter, dst_inter, E_UU, NU_, acc_u);
    // o6: uu_interby
    gemm(h_u, P.uu_interby.W, HID, HID, nullptr, hh_u, NU_, wbt0);
    gat(hh_u, NU_, hh_u, NU_, P.uu_interby.a_src, P.uu_interby.a_dst,
        src_interby, dst_interby, E_UU, NU_, acc_u);

    int relu = (l == 0) ? 1 : 0;
    hipLaunchKernelGGL(finalize3, dim3(cdiv((long)NT_ * HID, 256)), dim3(256), 0, stream,
                       acc_t, P.tt_replies.b, P.tt_replied.b, P.ut_posts.b, h_t, NT_, relu);
    hipLaunchKernelGGL(finalize3, dim3(cdiv((long)NU_ * HID, 256)), dim3(256), 0, stream,
                       acc_u, P.tu_posted.b, P.uu_inter.b, P.uu_interby.b, h_u, NU_, relu);
  }

  // -------- output heads, write straight into d_out (tweets then users) --------
  float* out_f = (float*)d_out;
  gemm(h_t, Wo, HID, 128, bo, out_f, NT_, wbt0);
  gemm(h_u, Wp, HID, 128, bp, out_f + (size_t)NT_ * 128, NU_, wbt0);
}
